// EfficientLinearAttention_39694087750160
// MI455X (gfx1250) — matrix operand order repaired and checked by reading, not compiled, not run on hardware
//
#include <hip/hip_runtime.h>
#include <hip/hip_bf16.h>

// ---------------------------------------------------------------------------
// EfficientLinearAttention for MI455X (gfx1250, wave32, WMMA bf16)
//
// Key optimizations:
//  * Support-prototype half of "joint" (batches B..2B-1) provably never
//    influences output[:B] -> skipped entirely (2x less work).
//  * All GEMMs on v_wmma_f32_16x16x32_bf16, f32 accumulate.
//  * Dual activation layouts ([ch][n] for K/V path, [n][ch] for X/Q/attended
//    path) make every WMMA fragment a contiguous per-lane global load:
//    no LDS staging, no per-K-step barriers -> compiler emits clause-packed
//    global_load_b128 pipelined against wmma (verified in round-1 disasm).
// ---------------------------------------------------------------------------

typedef __attribute__((ext_vector_type(16))) __bf16 v16bf;
typedef __attribute__((ext_vector_type(8)))  __bf16 bf16x8;
typedef __attribute__((ext_vector_type(4)))  __bf16 bf16x4;
typedef __attribute__((ext_vector_type(8)))  float  v8f;
typedef __attribute__((ext_vector_type(4)))  float  f32x4;

#define NB    8      // batches we actually compute
#define CC    512    // channels (C = KC = VC)
#define NN    4096   // H*W
#define NHEAD 8
#define DK    64
#define DV    64

__device__ __forceinline__ v8f v8f_zero() {
  v8f z = {0.f, 0.f, 0.f, 0.f, 0.f, 0.f, 0.f, 0.f};
  return z;
}

// A fragment (16x32, M=row): lane holds M=row, K halves {kk+g*8..+8} U {kk+16+g*8..+8}
__device__ __forceinline__ v16bf load_frag_a(const __bf16* __restrict__ row, int kk, int g) {
  bf16x8 lo = *(const bf16x8*)(row + kk + g * 8);
  bf16x8 hi = *(const bf16x8*)(row + kk + 16 + g * 8);
  v16bf a;
#pragma unroll
  for (int i = 0; i < 8; ++i) { a[i] = lo[i]; a[8 + i] = hi[i]; }
  return a;
}

// B fragment (32x16, N=row of transposed operand): lane holds N, K halves {kk+g*8..+8} U {kk+16+g*8..+8}
__device__ __forceinline__ v16bf load_frag_b(const __bf16* __restrict__ row, int kk, int g) {
  bf16x8 lo = *(const bf16x8*)(row + kk + g * 8);
  bf16x8 hi = *(const bf16x8*)(row + kk + 16 + g * 8);
  v16bf b;
#pragma unroll
  for (int i = 0; i < 8; ++i) { b[i] = lo[i]; b[8 + i] = hi[i]; }
  return b;
}

// --------------------------- f32 -> bf16 pack (weights) --------------------
__global__ __launch_bounds__(256)
void cvt_f32_bf16_kernel(const float* __restrict__ in, __bf16* __restrict__ out, int n) {
  int i = blockIdx.x * 256 + threadIdx.x;
  if (i < n) out[i] = (__bf16)in[i];
}

// --------------------------- [c][n] f32 -> [n][c] bf16 transpose -----------
__global__ __launch_bounds__(256)
void transpose_pack_kernel(const float* __restrict__ X,  // [b][c][n] f32
                           __bf16* __restrict__ XT) {    // [b][n][c] bf16
  __shared__ float tile[32][33];
  const int b = blockIdx.z;
  const int cTile = blockIdx.y * 32;
  const int nTile = blockIdx.x * 32;
  const int tid = threadIdx.x;
  const int row  = tid >> 3;        // 0..31
  const int col4 = (tid & 7) * 4;   // 0,4,..,28

  f32x4 v = *(const f32x4*)(X + ((size_t)b * CC + cTile + row) * NN + nTile + col4);
#pragma unroll
  for (int i = 0; i < 4; ++i) tile[row][col4 + i] = v[i];
  __syncthreads();

  bf16x4 o;
#pragma unroll
  for (int i = 0; i < 4; ++i) o[i] = (__bf16)tile[col4 + i][row];
  *(bf16x4*)(XT + ((size_t)b * NN + nTile + row) * CC + cTile + col4) = o;
}

// --------------------------- direct-global WMMA GEMM -----------------------
// Y[b][co][n] = sum_ci W[co][ci] * XT[b][n][ci] + bias[co]
// Block = 8 waves over a 128x128 tile; each wave computes 64x32 (4x2 tiles).
// All fragments are contiguous per-lane global loads (W row-major over ci;
// XT row-major over ci) -> no LDS, no barriers; L2 (192MB) holds W and XT.
// OUT_MODE 0: bf16 [co][n]   (K raw, V)
// OUT_MODE 1: bf16 [n][co]   (Q raw transposed; vector stores)
// OUT_MODE 2: f32  [co][n] + residual (final reproj -> d_out)
template <int OUT_MODE>
__global__ __launch_bounds__(256)
void gemm_direct_kernel(const __bf16* __restrict__ W,   // [512,512] bf16 (co,ci)
                        const __bf16* __restrict__ XT,  // [NB,4096,512] bf16
                        const float*  __restrict__ bias,
                        const float*  __restrict__ resid,
                        void* __restrict__ Yv) {
  const int tid = threadIdx.x, lane = tid & 31, w = tid >> 5;
  const int g = lane >> 4, ln = lane & 15;
  const int b = blockIdx.z;
  const int coTile = blockIdx.y * 128 + (w & 1) * 64;
  const int nTile  = blockIdx.x * 128 + (w >> 1) * 32;
  const __bf16* XTb = XT + (size_t)b * NN * CC;

  const __bf16* arow[4];
  const __bf16* brow[2];
#pragma unroll
  for (int mt = 0; mt < 4; ++mt) arow[mt] = W + (size_t)(coTile + mt * 16 + ln) * CC;
#pragma unroll
  for (int nt = 0; nt < 2; ++nt) brow[nt] = XTb + (size_t)(nTile + nt * 16 + ln) * CC;

  v8f acc[4][2];
#pragma unroll
  for (int mt = 0; mt < 4; ++mt)
#pragma unroll
    for (int nt = 0; nt < 2; ++nt) acc[mt][nt] = v8f_zero();

  for (int kk = 0; kk < CC; kk += 32) {
    if (kk + 32 < CC) {  // keep next K slab warm -> global_prefetch_b8
      __builtin_prefetch(arow[0] + kk + 32, 0, 1);
      __builtin_prefetch(brow[0] + kk + 32, 0, 1);
    }
    v16bf afr[4], bfr[2];
#pragma unroll
    for (int mt = 0; mt < 4; ++mt) afr[mt] = load_frag_a(arow[mt], kk, g);
#pragma unroll
    for (int nt = 0; nt < 2; ++nt) bfr[nt] = load_frag_b(brow[nt], kk, g);
#pragma unroll
    for (int mt = 0; mt < 4; ++mt)
#pragma unroll
      for (int nt = 0; nt < 2; ++nt)
        acc[mt][nt] = __builtin_amdgcn_wmma_f32_16x16x32_bf16(
            false, afr[mt], false, bfr[nt], (short)0, acc[mt][nt], false, false);
  }

  // C/D layout: (VGPR r, lane) -> M = r + 8*g (contiguous in r), N = ln
#pragma unroll
  for (int mt = 0; mt < 4; ++mt) {
#pragma unroll
    for (int nt = 0; nt < 2; ++nt) {
      const int n = nTile + nt * 16 + ln;
      if constexpr (OUT_MODE == 1) {
        const int co0 = coTile + mt * 16 + 8 * g;
        bf16x8 o;
#pragma unroll
        for (int r = 0; r < 8; ++r) o[r] = (__bf16)(acc[mt][nt][r] + bias[co0 + r]);
        *(bf16x8*)((__bf16*)Yv + ((size_t)b * NN + n) * CC + co0) = o;
      } else {
#pragma unroll
        for (int r = 0; r < 8; ++r) {
          const int co = coTile + mt * 16 + r + 8 * g;
          const size_t off = ((size_t)b * CC + co) * NN + n;
          float v = acc[mt][nt][r] + bias[co];
          if constexpr (OUT_MODE == 2) ((float*)Yv)[off] = v + resid[off];
          else                         ((__bf16*)Yv)[off] = (__bf16)v;
        }
      }
    }
  }
}

// --------------------------- spatial softmax (keys, over N=4096) -----------
__global__ __launch_bounds__(256)
void softmax_n_kernel(const __bf16* __restrict__ in, __bf16* __restrict__ out) {
  const int row = blockIdx.x;  // 0 .. NB*CC-1
  const __bf16* src = in + (size_t)row * NN;
  __bf16*       dst = out + (size_t)row * NN;
  const int tid = threadIdx.x;

  float vals[16];
  float m = -1e30f;
#pragma unroll
  for (int j = 0; j < 16; ++j) {
    vals[j] = (float)src[tid + j * 256];
    m = fmaxf(m, vals[j]);
  }
#pragma unroll
  for (int off = 16; off > 0; off >>= 1) m = fmaxf(m, __shfl_xor(m, off, 32));
  __shared__ float red[8];
  if ((tid & 31) == 0) red[tid >> 5] = m;
  __syncthreads();
  float bm = red[0];
#pragma unroll
  for (int i = 1; i < 8; ++i) bm = fmaxf(bm, red[i]);
  __syncthreads();

  float s = 0.f;
#pragma unroll
  for (int j = 0; j < 16; ++j) { vals[j] = __expf(vals[j] - bm); s += vals[j]; }
#pragma unroll
  for (int off = 16; off > 0; off >>= 1) s += __shfl_xor(s, off, 32);
  if ((tid & 31) == 0) red[tid >> 5] = s;
  __syncthreads();
  float bs = 0.f;
#pragma unroll
  for (int i = 0; i < 8; ++i) bs += red[i];
  const float inv = 1.0f / bs;
#pragma unroll
  for (int j = 0; j < 16; ++j) dst[tid + j * 256] = (__bf16)(vals[j] * inv);
}

// --------------------------- channel softmax (queries, over dk=64) ---------
// Operates on transposed layout [b][n][ch]: each thread's 64 values are a
// fully contiguous 128B run -> vector loads/stores instead of strided access.
__global__ __launch_bounds__(256)
void softmax_dk_kernel(const __bf16* __restrict__ inT, __bf16* __restrict__ outT) {
  const int n = blockIdx.x * 256 + threadIdx.x;
  const int h = blockIdx.y;
  const int b = blockIdx.z;
  const __bf16* src = inT + ((size_t)b * NN + n) * CC + h * DK;
  __bf16*       dst = outT + ((size_t)b * NN + n) * CC + h * DK;

  float v[64];
#pragma unroll
  for (int c8 = 0; c8 < 8; ++c8) {
    bf16x8 x = *(const bf16x8*)(src + c8 * 8);
#pragma unroll
    for (int i = 0; i < 8; ++i) v[c8 * 8 + i] = (float)x[i];
  }
  float m = -1e30f;
#pragma unroll
  for (int k = 0; k < 64; ++k) m = fmaxf(m, v[k]);
  float s = 0.f;
#pragma unroll
  for (int k = 0; k < 64; ++k) { v[k] = __expf(v[k] - m); s += v[k]; }
  const float inv = 1.0f / s;
#pragma unroll
  for (int c8 = 0; c8 < 8; ++c8) {
    bf16x8 o;
#pragma unroll
    for (int i = 0; i < 8; ++i) o[i] = (__bf16)(v[c8 * 8 + i] * inv);
    *(bf16x8*)(dst + c8 * 8) = o;
  }
}

// --------------------------- context = Ksm @ V^T  (per b,h: 64x64, K=4096) -
// Fragments load straight from global ([ch][n] layout). Output written
// transposed as bf16 (ctxT[v][k]) with one bf16x8 store per tile per lane,
// which is exactly the A-operand layout the attended kernel needs.
__global__ __launch_bounds__(256)
void context_kernel(const __bf16* __restrict__ Ksm, const __bf16* __restrict__ V,
                    __bf16* __restrict__ ctxT /* [b][h][v][k] bf16 */) {
  const int bh = blockIdx.x;  // 0..63
  const int b = bh >> 3, h = bh & 7;
  const int tid = threadIdx.x;
  const int lane = tid & 31, w = tid >> 5;
  const int g = lane >> 4, ln = lane & 15;

#pragma unroll
  for (int t = 0; t < 2; ++t) {
    const int tt = w * 2 + t;  // 16 tiles over 64x64
    const int mt = tt >> 2, vt = tt & 3;
    const __bf16* arow = Ksm + (size_t)(b * CC + h * DK + mt * 16 + ln) * NN;
    const __bf16* brow = V   + (size_t)(b * CC + h * DV + vt * 16 + ln) * NN;
    v8f acc = v8f_zero();
    for (int kk = 0; kk < NN; kk += 32) {
      v16bf a  = load_frag_a(arow, kk, g);
      v16bf bb = load_frag_b(brow, kk, g);
      acc = __builtin_amdgcn_wmma_f32_16x16x32_bf16(
          false, a, false, bb, (short)0, acc, false, false);
    }
    // lane has N = v = vt*16+ln fixed; M = k = mt*16 + 8g + r contiguous in r
    bf16x8 o;
#pragma unroll
    for (int r = 0; r < 8; ++r) o[r] = (__bf16)acc[r];
    *(bf16x8*)(ctxT + ((size_t)bh * DV + vt * 16 + ln) * DK + mt * 16 + 8 * g) = o;
  }
}

// --------------------------- attended = ctx^T @ Qsm  (per b,h: 64xN, K=64) -
// LDS-free: A from ctxT[v][k] (8KB per head, cache-resident), B from
// QsmT[n][k]. Output written [n][ch] bf16 (vector stores) feeding reproj.
__global__ __launch_bounds__(256)
void attended_kernel(const __bf16* __restrict__ ctxT, const __bf16* __restrict__ QsmT,
                     __bf16* __restrict__ attT /* [b][n][ch] */) {
  const int tid = threadIdx.x, lane = tid & 31, w = tid >> 5;
  const int g = lane >> 4, ln = lane & 15;
  const int nBase = blockIdx.x * 256 + w * 32;  // wave covers 32 pixels
  const int h = blockIdx.y, b = blockIdx.z;

  const __bf16* actx = ctxT + (size_t)(b * NHEAD + h) * DV * DK;
  const __bf16* arow[4];
  const __bf16* brow[2];
#pragma unroll
  for (int mt = 0; mt < 4; ++mt) arow[mt] = actx + (size_t)(mt * 16 + ln) * DK;
#pragma unroll
  for (int nt = 0; nt < 2; ++nt)
    brow[nt] = QsmT + ((size_t)b * NN + nBase + nt * 16 + ln) * CC + h * DK;

  v8f acc[4][2];
#pragma unroll
  for (int mt = 0; mt < 4; ++mt)
#pragma unroll
    for (int nt = 0; nt < 2; ++nt) acc[mt][nt] = v8f_zero();

#pragma unroll
  for (int kk = 0; kk < DK; kk += 32) {
    v16bf afr[4], bfr[2];
#pragma unroll
    for (int mt = 0; mt < 4; ++mt) afr[mt] = load_frag_a(arow[mt], kk, g);
#pragma unroll
    for (int nt = 0; nt < 2; ++nt) bfr[nt] = load_frag_b(brow[nt], kk, g);
#pragma unroll
    for (int mt = 0; mt < 4; ++mt)
#pragma unroll
      for (int nt = 0; nt < 2; ++nt)
        acc[mt][nt] = __builtin_amdgcn_wmma_f32_16x16x32_bf16(
            false, afr[mt], false, bfr[nt], (short)0, acc[mt][nt], false, false);
  }

#pragma unroll
  for (int mt = 0; mt < 4; ++mt) {
#pragma unroll
    for (int nt = 0; nt < 2; ++nt) {
      const int n = nBase + nt * 16 + ln;
      bf16x8 o;
#pragma unroll
      for (int r = 0; r < 8; ++r) o[r] = (__bf16)acc[mt][nt][r];
      *(bf16x8*)(attT + ((size_t)b * NN + n) * CC + h * DV + mt * 16 + 8 * g) = o;
    }
  }
}

// ---------------------------------------------------------------------------
extern "C" void kernel_launch(void* const* d_in, const int* in_sizes, int n_in,
                              void* d_out, int out_size, void* d_ws, size_t ws_size,
                              hipStream_t stream) {
  (void)in_sizes; (void)n_in; (void)out_size; (void)ws_size;

  const float* query_feat = (const float*)d_in[0];
  // d_in[1] = support_proto: provably unused for output[:B]
  const float* keys_w    = (const float*)d_in[2];
  const float* keys_b    = (const float*)d_in[3];
  const float* queries_w = (const float*)d_in[4];
  const float* queries_b = (const float*)d_in[5];
  const float* values_w  = (const float*)d_in[6];
  const float* values_b  = (const float*)d_in[7];
  const float* reproj_w  = (const float*)d_in[8];
  const float* reproj_b  = (const float*)d_in[9];

  const size_t WSZ = (size_t)CC * CC;        // 262144 weight elems
  const size_t MAT = (size_t)NB * CC * NN;   // 16.8M activation elems

  char* ws = (char*)d_ws;
  size_t off = 0;
  auto alloc = [&](size_t bytes) -> void* {
    void* p = ws + off;
    off += (bytes + 255) & ~(size_t)255;
    return p;
  };
  __bf16* Wk    = (__bf16*)alloc(WSZ * 2);
  __bf16* Wq    = (__bf16*)alloc(WSZ * 2);
  __bf16* Wv    = (__bf16*)alloc(WSZ * 2);
  __bf16* Wr    = (__bf16*)alloc(WSZ * 2);
  __bf16* XbT   = (__bf16*)alloc(MAT * 2);   // [b][n][c]
  __bf16* Kraw  = (__bf16*)alloc(MAT * 2);   // [b][c][n]
  __bf16* Ksm   = (__bf16*)alloc(MAT * 2);   // [b][c][n]
  __bf16* Vb    = (__bf16*)alloc(MAT * 2);   // [b][c][n]
  __bf16* QrawT = (__bf16*)alloc(MAT * 2);   // [b][n][c]
  __bf16* QsmT  = (__bf16*)alloc(MAT * 2);   // [b][n][c]
  __bf16* ctxT  = (__bf16*)alloc((size_t)NB * NHEAD * DV * DK * 2);
  __bf16* attT  = Kraw;  // alias: Kraw dead after softmax_n; layout [b][n][c]

  // pack weights, transpose+pack activations
  cvt_f32_bf16_kernel<<<(int)((WSZ + 255) / 256), 256, 0, stream>>>(keys_w,    Wk, (int)WSZ);
  cvt_f32_bf16_kernel<<<(int)((WSZ + 255) / 256), 256, 0, stream>>>(queries_w, Wq, (int)WSZ);
  cvt_f32_bf16_kernel<<<(int)((WSZ + 255) / 256), 256, 0, stream>>>(values_w,  Wv, (int)WSZ);
  cvt_f32_bf16_kernel<<<(int)((WSZ + 255) / 256), 256, 0, stream>>>(reproj_w,  Wr, (int)WSZ);
  transpose_pack_kernel<<<dim3(NN / 32, CC / 32, NB), 256, 0, stream>>>(query_feat, XbT);

  // K / Q / V projections (keys_b cancels in softmax but is free to include)
  dim3 gg(NN / 128, CC / 128, NB);  // (32, 4, 8)
  gemm_direct_kernel<0><<<gg, 256, 0, stream>>>(Wk, XbT, keys_b,    nullptr, Kraw);
  gemm_direct_kernel<1><<<gg, 256, 0, stream>>>(Wq, XbT, queries_b, nullptr, QrawT);
  gemm_direct_kernel<0><<<gg, 256, 0, stream>>>(Wv, XbT, values_b,  nullptr, Vb);

  // softmaxes
  softmax_n_kernel<<<NB * CC, 256, 0, stream>>>(Kraw, Ksm);
  softmax_dk_kernel<<<dim3(NN / 256, NHEAD, NB), 256, 0, stream>>>(QrawT, QsmT);

  // attention core
  context_kernel<<<NB * NHEAD, 256, 0, stream>>>(Ksm, Vb, ctxT);
  attended_kernel<<<dim3(NN / 256, NHEAD, NB), 256, 0, stream>>>(ctxT, QsmT, attT);

  // reproj + bias + residual -> f32 output
  gemm_direct_kernel<2><<<gg, 256, 0, stream>>>(Wr, attT, reproj_b, query_feat, d_out);
}
